// DifferentialAttentionLayer_2963527434481
// MI455X (gfx1250) — matrix operand order repaired and checked by reading, not compiled, not run on hardware
//
#include <hip/hip_runtime.h>
#include <hip/hip_bf16.h>
#include <math.h>

typedef _Float16 half16 __attribute__((ext_vector_type(16)));
typedef _Float16 half8  __attribute__((ext_vector_type(8)));
typedef float    floatx8 __attribute__((ext_vector_type(8)));
typedef float    floatx4 __attribute__((ext_vector_type(4)));

#define EMBED_DIM 1024
#define NHEAD 16
#define HD 64
#define HHD 32
#define SEQ_B 2
#define SEQ_T 2048
#define MTOK (SEQ_B*SEQ_T)   /* 4096 tokens */
#define LAMBDA_INIT 0.8f
#define LN_EPS 1e-5f

union H16U { half16 v; half8 h[2]; };

// Build a 16-half WMMA fragment from two 16-byte chunks (2x b128 loads).
static __device__ inline half16 ld_frag(const _Float16* p0, const _Float16* p1) {
  H16U u;
  u.h[0] = *(const half8*)p0;
  u.h[1] = *(const half8*)p1;
  return u.v;
}

static __device__ inline floatx8 wmma_f16(half16 a, half16 b, floatx8 c) {
  return __builtin_amdgcn_wmma_f32_16x16x32_f16(false, a, false, b, (short)0, c,
                                                false, false);
}

// CDNA5 async global->LDS copy (ASYNCcnt-tracked), 16 bytes per lane.
// lds_off is the raw LDS byte address (single __shared__ object => base 0).
static __device__ inline void async_copy_b128(unsigned lds_off,
                                              const _Float16* gsrc) {
  unsigned long long ga = (unsigned long long)(uintptr_t)gsrc;
  asm volatile("global_load_async_to_lds_b128 %0, %1, off"
               :: "v"(lds_off), "v"(ga) : "memory");
}
static __device__ inline void wait_async_all() {
  asm volatile("s_wait_asynccnt 0x0" ::: "memory");
}

// ---------------------------------------------------------------------------
// Tiled GEMM: C[M,N] = A[M,K] @ W[N,K]^T
//   IN_F16:    A is f16 (staged to LDS with async-to-LDS DMA)
//              else f32 (converted on stage-in)
//   TRANS_OUT: store C as f16 [N, M] (used for V^T)
//   RESID:     C = A@W^T + resid, stored f32 [M, N]
// Block: 256 threads = 8 waves; block tile 128(M) x 64(N); k-chunks of 64.
// ---------------------------------------------------------------------------
template<bool IN_F16, bool TRANS_OUT, bool RESID>
__global__ __launch_bounds__(256)
void gemm16(const void* __restrict__ Ain, const float* __restrict__ W,
            const float* __restrict__ resid,
            _Float16* __restrict__ Ch, float* __restrict__ Cf,
            int M, int N, int K)
{
  // Single shared object so sA sits at LDS byte offset 0 (needed for the
  // raw LDS addresses fed to global_load_async_to_lds_b128).
  __shared__ __align__(16) _Float16 smem[128 * 64 + 64 * 64];
  _Float16* sA = smem;             // [128][64]
  _Float16* sB = smem + 128 * 64;  // [64][64]

  const int tid  = threadIdx.x;
  const int lane = tid & 31;
  const int wave = tid >> 5;
  const int hl   = lane >> 4;   // which 16-lane half
  const int l16  = lane & 15;
  const int m0   = blockIdx.y * 128;
  const int n0   = blockIdx.x * 64;

  floatx8 acc[4] = {};

  for (int kc = 0; kc < K; kc += 64) {
    const int r  = tid >> 1;      // 0..127
    const int kh = tid & 1;       // which 32-wide half of the k-chunk
    // ---- stage A tile: 128 rows x 64 k ----
    if (IN_F16) {
      const _Float16* src =
          (const _Float16*)Ain + (size_t)(m0 + r) * K + kc + kh * 32;
      const unsigned base = (unsigned)((r * 64 + kh * 32) * 2);  // bytes
      async_copy_b128(base + 0,  src + 0);
      async_copy_b128(base + 16, src + 8);
      async_copy_b128(base + 32, src + 16);
      async_copy_b128(base + 48, src + 24);
    } else {
      const float* src = (const float*)Ain + (size_t)(m0 + r) * K + kc + kh * 32;
      _Float16* dst = &sA[r * 64 + kh * 32];
#pragma unroll
      for (int q = 0; q < 2; ++q) {
        floatx4 f0 = *(const floatx4*)(src + q * 16 + 0);
        floatx4 f1 = *(const floatx4*)(src + q * 16 + 4);
        floatx4 f2 = *(const floatx4*)(src + q * 16 + 8);
        floatx4 f3 = *(const floatx4*)(src + q * 16 + 12);
        half8 h0, h1;
#pragma unroll
        for (int i = 0; i < 4; ++i) {
          h0[i]     = (_Float16)f0[i];
          h0[i + 4] = (_Float16)f1[i];
          h1[i]     = (_Float16)f2[i];
          h1[i + 4] = (_Float16)f3[i];
        }
        *(half8*)(dst + q * 16)     = h0;
        *(half8*)(dst + q * 16 + 8) = h1;
      }
    }
    // ---- stage W tile: 64 rows x 64 k (f32 -> f16) ----
    if (tid < 128) {
      const int rw = tid >> 1;
      const float* src = W + (size_t)(n0 + rw) * K + kc + kh * 32;
      _Float16* dst = &sB[rw * 64 + kh * 32];
#pragma unroll
      for (int q = 0; q < 2; ++q) {
        floatx4 f0 = *(const floatx4*)(src + q * 16 + 0);
        floatx4 f1 = *(const floatx4*)(src + q * 16 + 4);
        floatx4 f2 = *(const floatx4*)(src + q * 16 + 8);
        floatx4 f3 = *(const floatx4*)(src + q * 16 + 12);
        half8 h0, h1;
#pragma unroll
        for (int i = 0; i < 4; ++i) {
          h0[i]     = (_Float16)f0[i];
          h0[i + 4] = (_Float16)f1[i];
          h1[i]     = (_Float16)f2[i];
          h1[i + 4] = (_Float16)f3[i];
        }
        *(half8*)(dst + q * 16)     = h0;
        *(half8*)(dst + q * 16 + 8) = h1;
      }
    }
    if (IN_F16) wait_async_all();   // ASYNCcnt drained before the barrier
    __syncthreads();

    // A fragments: row m = wave*16 + l16; chunk g: k = g*32 .. g*32+31
    const _Float16* arow = &sA[(wave * 16 + l16) * 64];
    const half16 a0 = ld_frag(arow + hl * 8,      arow + 16 + hl * 8);
    const half16 a1 = ld_frag(arow + 32 + hl * 8, arow + 48 + hl * 8);

#pragma unroll
    for (int tn = 0; tn < 4; ++tn) {
      // B fragments: n = tn*16 + l16; k = g*32 + table order (hl*8, 16+hl*8)
      const _Float16* brow = &sB[(tn * 16 + l16) * 64];
      const half16 b0 = ld_frag(brow + hl * 8,      brow + 16 + hl * 8);
      const half16 b1 = ld_frag(brow + 32 + hl * 8, brow + 48 + hl * 8);
      acc[tn] = wmma_f16(a0, b0, acc[tn]);
      acc[tn] = wmma_f16(a1, b1, acc[tn]);
    }
    __syncthreads();
  }

  // ---- epilogue: C layout lane holds n = l16, rows m = v + 8*hl ----
#pragma unroll
  for (int tn = 0; tn < 4; ++tn) {
    const int n     = n0 + tn * 16 + l16;
    const int mbase = m0 + wave * 16 + 8 * hl;
    if (TRANS_OUT) {
      half8 hv;
#pragma unroll
      for (int v = 0; v < 8; ++v) hv[v] = (_Float16)acc[tn][v];
      *(half8*)(Ch + (size_t)n * M + mbase) = hv;  // contiguous in m
    } else if (RESID) {
#pragma unroll
      for (int v = 0; v < 8; ++v) {
        const size_t idx = (size_t)(mbase + v) * N + n;
        Cf[idx] = acc[tn][v] + resid[idx];
      }
    } else {
#pragma unroll
      for (int v = 0; v < 8; ++v)
        Ch[(size_t)(mbase + v) * N + n] = (_Float16)acc[tn][v];
    }
  }
}

// ---------------------------------------------------------------------------
// Differential flash attention: one wave per (b, h, 16-query tile).
// Two independent online-softmax streams; 32 keys per iteration.
// Q,K layout: f16 [token][1024];  V layout: f16 transposed [feature][token].
// ---------------------------------------------------------------------------
__global__ __launch_bounds__(32)
void diff_attn(const _Float16* __restrict__ Q,
               const _Float16* __restrict__ Kb,
               const _Float16* __restrict__ VT,
               const unsigned char* __restrict__ kpm,
               const float* __restrict__ lq1, const float* __restrict__ lk1,
               const float* __restrict__ lq2, const float* __restrict__ lk2,
               _Float16* __restrict__ Out)
{
  __shared__ __align__(16) _Float16 sP1[16 * 32];
  __shared__ __align__(16) _Float16 sP2[16 * 32];

  const int lane = threadIdx.x;
  const int hl   = lane >> 4;
  const int l16  = lane & 15;
  const int qt   = blockIdx.x;
  const int h    = blockIdx.y;
  const int b    = blockIdx.z;
  const int t0   = qt * 16;

  // ---- lambda for this head: exp(lq1.lk1) - exp(lq2.lk2) + lambda_init ----
  float pp1 = lq1[h * HHD + lane] * lk1[h * HHD + lane];
  float pp2 = lq2[h * HHD + lane] * lk2[h * HHD + lane];
#pragma unroll
  for (int mask = 16; mask >= 1; mask >>= 1) {
    pp1 += __shfl_xor(pp1, mask, 32);
    pp2 += __shfl_xor(pp2, mask, 32);
  }
  const float lam = __expf(pp1) - __expf(pp2) + LAMBDA_INIT;

  // ---- Q fragments (A matrices), query row m = l16 ----
  const _Float16* qrow =
      Q + (size_t)(b * SEQ_T + t0 + l16) * EMBED_DIM + h * HD;
  const half16 aq1 = ld_frag(qrow + hl * 8,      qrow + 16 + hl * 8);
  const half16 aq2 = ld_frag(qrow + 32 + hl * 8, qrow + 48 + hl * 8);

  floatx8 O1[4] = {};
  floatx8 O2[4] = {};
  float m1[8], s1[8], m2[8], s2[8];
#pragma unroll
  for (int v = 0; v < 8; ++v) { m1[v] = -1e30f; m2[v] = -1e30f; s1[v] = 0.f; s2[v] = 0.f; }

  const float rscale = 0.17677669529663687f;  // 1/sqrt(HHD)

  for (int c = 0; c < SEQ_T; c += 32) {
    // ---- K fragments (B matrices): n = key, k = head-dim (table order) ----
    const _Float16* krowA =
        Kb + (size_t)(b * SEQ_T + c + l16) * EMBED_DIM + h * HD + hl * 8;
    const _Float16* krowB =
        Kb + (size_t)(b * SEQ_T + c + 16 + l16) * EMBED_DIM + h * HD + hl * 8;
    if (c + 32 < SEQ_T) __builtin_prefetch(krowA + 32 * EMBED_DIM, 0, 0);

    const half16 bk1a = ld_frag(krowA,      krowA + 16);
    const half16 bk1b = ld_frag(krowB,      krowB + 16);
    const half16 bk2a = ld_frag(krowA + 32, krowA + 48);
    const half16 bk2b = ld_frag(krowB + 32, krowB + 48);

    const floatx8 zero8 = {};
    floatx8 t1a = wmma_f16(aq1, bk1a, zero8);
    floatx8 t1b = wmma_f16(aq1, bk1b, zero8);
    floatx8 t2a = wmma_f16(aq2, bk2a, zero8);
    floatx8 t2b = wmma_f16(aq2, bk2b, zero8);

    // ---- scale + key-padding mask (per-lane key = c [+16] + l16) ----
    const bool ma = kpm[b * SEQ_T + c + l16] != 0;
    const bool mb = kpm[b * SEQ_T + c + 16 + l16] != 0;
#pragma unroll
    for (int v = 0; v < 8; ++v) {
      t1a[v] = ma ? -1e30f : t1a[v] * rscale;
      t1b[v] = mb ? -1e30f : t1b[v] * rscale;
      t2a[v] = ma ? -1e30f : t2a[v] * rscale;
      t2b[v] = mb ? -1e30f : t2b[v] * rscale;
    }

    // ---- online softmax, stream 1 (rows live across 16-lane halves) ----
#pragma unroll
    for (int v = 0; v < 8; ++v) {
      float mloc = fmaxf(t1a[v], t1b[v]);
      mloc = fmaxf(mloc, __shfl_xor(mloc, 1, 32));
      mloc = fmaxf(mloc, __shfl_xor(mloc, 2, 32));
      mloc = fmaxf(mloc, __shfl_xor(mloc, 4, 32));
      mloc = fmaxf(mloc, __shfl_xor(mloc, 8, 32));
      const float mnew = fmaxf(m1[v], mloc);
      const float corr = __expf(m1[v] - mnew);
      m1[v] = mnew;
      t1a[v] = __expf(t1a[v] - mnew);
      t1b[v] = __expf(t1b[v] - mnew);
      float sl = t1a[v] + t1b[v];
      sl += __shfl_xor(sl, 1, 32);
      sl += __shfl_xor(sl, 2, 32);
      sl += __shfl_xor(sl, 4, 32);
      sl += __shfl_xor(sl, 8, 32);
      s1[v] = s1[v] * corr + sl;
      O1[0][v] *= corr; O1[1][v] *= corr; O1[2][v] *= corr; O1[3][v] *= corr;
    }
    // ---- online softmax, stream 2 ----
#pragma unroll
    for (int v = 0; v < 8; ++v) {
      float mloc = fmaxf(t2a[v], t2b[v]);
      mloc = fmaxf(mloc, __shfl_xor(mloc, 1, 32));
      mloc = fmaxf(mloc, __shfl_xor(mloc, 2, 32));
      mloc = fmaxf(mloc, __shfl_xor(mloc, 4, 32));
      mloc = fmaxf(mloc, __shfl_xor(mloc, 8, 32));
      const float mnew = fmaxf(m2[v], mloc);
      const float corr = __expf(m2[v] - mnew);
      m2[v] = mnew;
      t2a[v] = __expf(t2a[v] - mnew);
      t2b[v] = __expf(t2b[v] - mnew);
      float sl = t2a[v] + t2b[v];
      sl += __shfl_xor(sl, 1, 32);
      sl += __shfl_xor(sl, 2, 32);
      sl += __shfl_xor(sl, 4, 32);
      sl += __shfl_xor(sl, 8, 32);
      s2[v] = s2[v] * corr + sl;
      O2[0][v] *= corr; O2[1][v] *= corr; O2[2][v] *= corr; O2[3][v] *= corr;
    }

    // ---- C-layout -> A-layout conversion of P through LDS ----
    __syncthreads();  // WAR vs previous iteration's fragment reads
#pragma unroll
    for (int v = 0; v < 8; ++v) {
      const int row = v + 8 * hl;
      sP1[row * 32 + l16]      = (_Float16)t1a[v];
      sP1[row * 32 + 16 + l16] = (_Float16)t1b[v];
      sP2[row * 32 + l16]      = (_Float16)t2a[v];
      sP2[row * 32 + 16 + l16] = (_Float16)t2b[v];
    }
    __syncthreads();
    const _Float16* prow1 = &sP1[l16 * 32];
    const _Float16* prow2 = &sP2[l16 * 32];
    const half16 ap1 = ld_frag(prow1 + hl * 8, prow1 + 16 + hl * 8);
    const half16 ap2 = ld_frag(prow2 + hl * 8, prow2 + 16 + hl * 8);

    // ---- accumulate P @ V (V^T layout makes B-frags contiguous) ----
#pragma unroll
    for (int tn = 0; tn < 4; ++tn) {
      const _Float16* vrow =
          VT + (size_t)(h * HD + tn * 16 + l16) * MTOK + b * SEQ_T + c + hl * 8;
      const half16 bv = ld_frag(vrow, vrow + 16);
      O1[tn] = wmma_f16(ap1, bv, O1[tn]);
      O2[tn] = wmma_f16(ap2, bv, O2[tn]);
    }
  }

  // ---- combine: out = O1/s1 - lam * O2/s2 ; store f16 [token][feature] ----
  float inv1[8], inv2[8];
#pragma unroll
  for (int v = 0; v < 8; ++v) { inv1[v] = 1.f / s1[v]; inv2[v] = lam / s2[v]; }
#pragma unroll
  for (int tn = 0; tn < 4; ++tn) {
#pragma unroll
    for (int v = 0; v < 8; ++v) {
      const float o = O1[tn][v] * inv1[v] - O2[tn][v] * inv2[v];
      Out[(size_t)(b * SEQ_T + t0 + v + 8 * hl) * EMBED_DIM
          + h * HD + tn * 16 + l16] = (_Float16)o;
    }
  }
}

// ---------------------------------------------------------------------------
// Row-wise LayerNorm over d=1024, one block (256 threads) per token row.
// ---------------------------------------------------------------------------
__global__ __launch_bounds__(256)
void layernorm_k(const float* __restrict__ Y, const float* __restrict__ gamma,
                 const float* __restrict__ beta, float* __restrict__ Out)
{
  __shared__ float red[16];
  const int row = blockIdx.x;
  const int tid = threadIdx.x;
  const float* y = Y + (size_t)row * EMBED_DIM;

  const floatx4 vals = *(const floatx4*)(y + tid * 4);
  float s  = vals[0] + vals[1] + vals[2] + vals[3];
  float s2 = vals[0] * vals[0] + vals[1] * vals[1] +
             vals[2] * vals[2] + vals[3] * vals[3];
#pragma unroll
  for (int mask = 16; mask >= 1; mask >>= 1) {
    s  += __shfl_xor(s, mask, 32);
    s2 += __shfl_xor(s2, mask, 32);
  }
  const int wave = tid >> 5, lane = tid & 31;
  if (lane == 0) { red[wave] = s; red[8 + wave] = s2; }
  __syncthreads();
  float ts = 0.f, ts2 = 0.f;
#pragma unroll
  for (int i = 0; i < 8; ++i) { ts += red[i]; ts2 += red[8 + i]; }
  const float mu   = ts / (float)EMBED_DIM;
  const float var  = ts2 / (float)EMBED_DIM - mu * mu;
  const float rstd = rsqrtf(var + LN_EPS);
#pragma unroll
  for (int i = 0; i < 4; ++i) {
    const int idx = tid * 4 + i;
    Out[(size_t)row * EMBED_DIM + idx] =
        (vals[i] - mu) * rstd * gamma[idx] + beta[idx];
  }
}

// ---------------------------------------------------------------------------
extern "C" void kernel_launch(void* const* d_in, const int* in_sizes, int n_in,
                              void* d_out, int out_size, void* d_ws, size_t ws_size,
                              hipStream_t stream)
{
  (void)in_sizes; (void)n_in; (void)out_size; (void)ws_size;

  const float*         x     = (const float*)d_in[0];
  const unsigned char* kpm   = (const unsigned char*)d_in[1];
  const float*         Wq    = (const float*)d_in[2];
  const float*         Wk    = (const float*)d_in[3];
  const float*         Wv    = (const float*)d_in[4];
  const float*         Wo    = (const float*)d_in[5];
  const float*         gamma = (const float*)d_in[6];
  const float*         beta  = (const float*)d_in[7];
  const float*         lq1   = (const float*)d_in[8];
  const float*         lk1   = (const float*)d_in[9];
  const float*         lq2   = (const float*)d_in[10];
  const float*         lk2   = (const float*)d_in[11];

  char* ws = (char*)d_ws;
  const size_t szH = (size_t)MTOK * EMBED_DIM * sizeof(_Float16);  // 8 MB
  _Float16* qbuf = (_Float16*)ws;  ws += szH;
  _Float16* kbuf = (_Float16*)ws;  ws += szH;
  _Float16* vT   = (_Float16*)ws;  ws += szH;   // transposed [feature][token]
  _Float16* abuf = (_Float16*)ws;  ws += szH;
  float*    ybuf = (float*)ws;                  // 16 MB

  const dim3 gB(EMBED_DIM / 64, MTOK / 128);
  gemm16<false, false, false><<<gB, 256, 0, stream>>>(
      x, Wq, nullptr, qbuf, nullptr, MTOK, EMBED_DIM, EMBED_DIM);
  gemm16<false, false, false><<<gB, 256, 0, stream>>>(
      x, Wk, nullptr, kbuf, nullptr, MTOK, EMBED_DIM, EMBED_DIM);
  gemm16<false, true, false><<<gB, 256, 0, stream>>>(
      x, Wv, nullptr, vT, nullptr, MTOK, EMBED_DIM, EMBED_DIM);

  const dim3 gA(SEQ_T / 16, NHEAD, SEQ_B);
  diff_attn<<<gA, 32, 0, stream>>>(qbuf, kbuf, vT, kpm, lq1, lk1, lq2, lk2, abuf);

  gemm16<true, false, true><<<gB, 256, 0, stream>>>(
      abuf, Wo, x, nullptr, ybuf, MTOK, EMBED_DIM, EMBED_DIM);

  layernorm_k<<<MTOK, 256, 0, stream>>>(ybuf, gamma, beta, (float*)d_out);
}
